// DECO_ConvNet_17377437680481
// MI455X (gfx1250) — compile-verified, matrix operand order checked by reading, not run on hardware
//
#include <hip/hip_runtime.h>
#include <hip/hip_bf16.h>
#include <math.h>

#define BATCHN 4
#define SEQN   256
#define MTOK   (BATCHN*SEQN)   // 1024 tokens everywhere

typedef _Float16 v16h __attribute__((ext_vector_type(16)));
typedef _Float16 h8   __attribute__((ext_vector_type(8)));
typedef float    v8f  __attribute__((ext_vector_type(8)));
typedef float    f4   __attribute__((ext_vector_type(4)));

enum { ACT_NONE = 0, ACT_GELU = 1 };

__device__ __forceinline__ float geluf(float x) { return 0.5f * x * (1.0f + erff(x * 0.70710678118654752f)); }
__device__ __forceinline__ float siluf(float x) { return x / (1.0f + __expf(-x)); }
__device__ __forceinline__ float softplusf(float x) { return (x > 20.0f) ? x : log1pf(__expf(x)); }

__device__ __forceinline__ v16h mk16(h8 lo, h8 hi) {
  v16h r;
#pragma unroll
  for (int i = 0; i < 8; i++) { r[i] = lo[i]; r[i + 8] = hi[i]; }
  return r;
}

// ---------------------------------------------------------------------------
// WMMA GEMM:  C(M=1024, N) = act( A(M,K; lda) @ W(N,K)^T + bias ) [* maskTok]
// f16 inputs staged in double-buffered LDS, f32 accumulate.
// Requires K % 32 == 0, M % 128 == 0.
// Block = 128 threads (4 wave32), 128x64 tile; each wave owns a 32x64 strip
// (2 A-frags x 4 B-frags = 8 v_wmma per 32-k step).
// ---------------------------------------------------------------------------
__global__ __launch_bounds__(128) void gemm_f16_wmma(
    const float* __restrict__ A, int lda,
    const float* __restrict__ W, int K,
    const float* __restrict__ bias,
    const float* __restrict__ maskTok,
    float* __restrict__ C, int ldc,
    int N, int act)
{
  __shared__ _Float16 As[2][128 * 32] __attribute__((aligned(16)));
  __shared__ _Float16 Bs[2][64 * 32]  __attribute__((aligned(16)));
  const int tid  = threadIdx.x;
  const int bm   = blockIdx.y << 7, bn = blockIdx.x << 6;
  const int wave = tid >> 5, lane = tid & 31;
  const int wm   = wave << 5;            // 4 waves cover 128 rows
  const int lr   = lane & 15, kh = lane >> 4;
  const int rowB = tid >> 1, kb = (tid & 1) << 4;   // B staging: half-row each

  v8f acc[2][4] = {};

  auto stage = [&](int buf, int k0) {
    { // A tile: each thread converts one full 32-k row (8x f4 -> 2x 16 f16)
      const float* s = A + (size_t)(bm + tid) * lda + k0;
      _Float16* d = &As[buf][tid * 32];
#pragma unroll
      for (int i = 0; i < 32; i += 4) {
        f4 v = *(const f4*)(s + i);
        d[i] = (_Float16)v[0]; d[i + 1] = (_Float16)v[1];
        d[i + 2] = (_Float16)v[2]; d[i + 3] = (_Float16)v[3];
      }
      if (k0 + 64 < K) __builtin_prefetch(s + 64, 0, 1);
    }
    { // W tile (64 n-rows x 32 k) with N guard
      int n = bn + rowB;
      _Float16* d = &Bs[buf][rowB * 32 + kb];
      if (n < N) {
        const float* s = W + (size_t)n * K + k0 + kb;
#pragma unroll
        for (int i = 0; i < 16; i += 4) {
          f4 v = *(const f4*)(s + i);
          d[i] = (_Float16)v[0]; d[i + 1] = (_Float16)v[1];
          d[i + 2] = (_Float16)v[2]; d[i + 3] = (_Float16)v[3];
        }
        if (k0 + 64 < K) __builtin_prefetch(s + 64, 0, 1);
      } else {
#pragma unroll
        for (int i = 0; i < 16; i++) d[i] = (_Float16)0.0f;
      }
    }
  };

  stage(0, 0);
  const int nk = K >> 5;
  int buf = 0;
  for (int kt = 0; kt < nk; kt++) {
    __syncthreads();                       // staging of `buf` visible; prev compute done
    if (kt + 1 < nk) stage(buf ^ 1, (kt + 1) << 5);

    v16h af[2], bf[4];
#pragma unroll
    for (int i = 0; i < 2; i++) {
      const _Float16* p = &As[buf][(wm + i * 16 + lr) * 32];
      af[i] = mk16(*(const h8*)(p + kh * 8), *(const h8*)(p + 16 + kh * 8));
    }
#pragma unroll
    for (int j = 0; j < 4; j++) {
      const _Float16* q = &Bs[buf][(j * 16 + lr) * 32];
      bf[j] = mk16(*(const h8*)(q + kh * 8), *(const h8*)(q + 16 + kh * 8));
    }
#pragma unroll
    for (int i = 0; i < 2; i++)
#pragma unroll
      for (int j = 0; j < 4; j++)
        acc[i][j] = __builtin_amdgcn_wmma_f32_16x16x32_f16(
            false, af[i], false, bf[j], (short)0, acc[i][j], false, false);
    buf ^= 1;
  }

#pragma unroll
  for (int i = 0; i < 2; i++)
#pragma unroll
    for (int j = 0; j < 4; j++)
#pragma unroll
      for (int r = 0; r < 8; r++) {
        int m = bm + wm + i * 16 + kh * 8 + r;
        int n = bn + j * 16 + lr;
        if (n < N) {
          float v = acc[i][j][r];
          if (bias)    v += bias[n];
          if (act == ACT_GELU) v = geluf(v);
          if (maskTok) v *= maskTok[m];
          C[(size_t)m * ldc + n] = v;
        }
      }
}

// ---------------------------------------------------------------------------
// Depthwise conv over L in (B,L,C) layout. y[t,c] = sum_k x[t-padL+k, c]*w[c,k]
// Optional SiLU, optional per-token mask multiply.
// ---------------------------------------------------------------------------
__global__ void dwconv_kernel(const float* __restrict__ X, int ldx,
                              const float* __restrict__ Wt, const float* __restrict__ Bv,
                              const float* __restrict__ mk,
                              float* __restrict__ Y, int ldy,
                              int L, int C, int taps, int padL, int silu, int total)
{
  int idx = blockIdx.x * blockDim.x + threadIdx.x;
  if (idx >= total) return;
  int c = idx % C, t = idx / C, l = t % L, tb = t - l;
  float acc = Bv ? Bv[c] : 0.0f;
  for (int k = 0; k < taps; k++) {
    int li = l - padL + k;
    if (li >= 0 && li < L) acc += X[(size_t)(tb + li) * ldx + c] * Wt[c * taps + k];
  }
  if (silu) acc = siluf(acc);
  if (mk)   acc *= mk[t];
  Y[(size_t)t * ldy + c] = acc;
}

// LayerNorm over channel dim, one block per token (C in {512,1024}).
__global__ __launch_bounds__(256) void ln_kernel(const float* __restrict__ X, int ldx,
                                                 const float* __restrict__ w, const float* __restrict__ b,
                                                 float* __restrict__ Y, int ldy, int C, float eps)
{
  __shared__ float red[256];
  int t = blockIdx.x;
  const float* x = X + (size_t)t * ldx;
  int nv = C >> 8;
  float v[4];
  float s = 0.0f;
  for (int i = 0; i < nv; i++) { v[i] = x[threadIdx.x + (i << 8)]; s += v[i]; }
  red[threadIdx.x] = s; __syncthreads();
  for (int o = 128; o > 0; o >>= 1) { if (threadIdx.x < o) red[threadIdx.x] += red[threadIdx.x + o]; __syncthreads(); }
  float mu = red[0] / (float)C; __syncthreads();
  float s2 = 0.0f;
  for (int i = 0; i < nv; i++) { float d = v[i] - mu; s2 += d * d; }
  red[threadIdx.x] = s2; __syncthreads();
  for (int o = 128; o > 0; o >>= 1) { if (threadIdx.x < o) red[threadIdx.x] += red[threadIdx.x + o]; __syncthreads(); }
  float rs = rsqrtf(red[0] / (float)C + eps);
  for (int i = 0; i < nv; i++) {
    int c = threadIdx.x + (i << 8);
    Y[(size_t)t * ldy + c] = (v[i] - mu) * rs * w[c] + b[c];
  }
}

// dt projection: dt = softplus(xdbl[:, :8] @ dtw(256,8)^T + dtb), K=8 (VALU).
__global__ __launch_bounds__(256) void dtproj_kernel(const float* __restrict__ xdbl,
                                                     const float* __restrict__ dtw,
                                                     const float* __restrict__ dtb,
                                                     float* __restrict__ dt)
{
  __shared__ float xs[8];
  int m = blockIdx.x, n = threadIdx.x;
  if (n < 8) xs[n] = xdbl[(size_t)m * 40 + n];
  __syncthreads();
  float a = dtb[n];
#pragma unroll
  for (int k = 0; k < 8; k++) a += xs[k] * dtw[n * 8 + k];
  dt[(size_t)m * 256 + n] = softplusf(a);
}

// Selective scan: block per batch, thread per channel d (256), state h[16] in regs.
__global__ __launch_bounds__(256) void scan_kernel(const float* __restrict__ dt,
                                                   const float* __restrict__ xdbl,
                                                   const float* __restrict__ xconv,
                                                   const float* __restrict__ z, int ldz,
                                                   const float* __restrict__ A_log,
                                                   const float* __restrict__ Dp,
                                                   float* __restrict__ Y, int L)
{
  int b = blockIdx.x, d = threadIdx.x;
  float Aa[16], h[16];
#pragma unroll
  for (int s = 0; s < 16; s++) { Aa[s] = -__expf(A_log[d * 16 + s]); h[s] = 0.0f; }
  float Dd = Dp[d];
  for (int l = 0; l < L; l++) {
    size_t t = (size_t)(b * L + l);
    float dtv = dt[t * 256 + d];
    float xv  = xconv[t * 256 + d];
    const float* bc = &xdbl[t * 40 + 8];
    const float* cc = &xdbl[t * 40 + 24];
    float acc = 0.0f;
#pragma unroll
    for (int s = 0; s < 16; s++) {
      float dA = __expf(dtv * Aa[s]);
      h[s] = dA * h[s] + dtv * bc[s] * xv;
      acc += h[s] * cc[s];
    }
    float zv = z[t * ldz + d];
    float y = acc + Dd * xv;
    y *= zv / (1.0f + __expf(-zv));
    Y[t * 256 + d] = y;
  }
}

// ------------------------- small elementwise helpers -------------------------
__global__ void add_kernel(const float* a, const float* b, float* o, int n) {
  int i = blockIdx.x * blockDim.x + threadIdx.x; if (i < n) o[i] = a[i] + b[i];
}
__global__ void resg_kernel(const float* res, const float* g, const float* y, float* o, int n, int C) {
  int i = blockIdx.x * blockDim.x + threadIdx.x; if (i < n) o[i] = res[i] + g[i % C] * y[i];
}
__global__ void copy_kernel(const float* a, float* o, int n) {
  int i = blockIdx.x * blockDim.x + threadIdx.x; if (i < n) o[i] = a[i];
}
__global__ void maskf_kernel(const unsigned char* m, float* o, int n) {
  int i = blockIdx.x * blockDim.x + threadIdx.x; if (i < n) o[i] = m[i] ? 1.0f : 0.0f;
}
__global__ void t_bcl_blc(const float* in, float* out, int C, int L, int total) {
  int i = blockIdx.x * blockDim.x + threadIdx.x; if (i >= total) return;
  int c = i % C, t = i / C, l = t % L, b = t / L;
  out[i] = in[((size_t)b * C + c) * L + l];
}
__global__ void t_blc_bcl(const float* in, float* out, int C, int L, int total) {
  int i = blockIdx.x * blockDim.x + threadIdx.x; if (i >= total) return;
  int l = i % L, t = i / L, c = t % C, b = t / C;
  out[i] = in[((size_t)b * L + l) * C + c];
}
__global__ void bcast_kernel(const float* te, float* o, int n, int chunk) {
  int i = blockIdx.x * blockDim.x + threadIdx.x; if (i < n) o[i] = te[i % chunk];
}

// ---------------------------------------------------------------------------
extern "C" void kernel_launch(void* const* d_in, const int* in_sizes, int n_in,
                              void* d_out, int out_size, void* d_ws, size_t ws_size,
                              hipStream_t stream)
{
  (void)in_sizes; (void)n_in; (void)out_size; (void)ws_size;
  const float* src_bcl  = (const float*)d_in[0];
  const float* qpos_bcl = (const float*)d_in[1];
  const unsigned char* maskb = (const unsigned char*)d_in[2];

  // ---- parse params (jax tree-flatten order: dicts alphabetical) ----
  int idx = 3;
  auto nxt = [&]() -> const float* { return (const float*)d_in[idx++]; };
  struct MambaP { const float *A_log,*D,*conv_b,*conv_w,*dt_b,*dt_w,*in_w,*out_w,*xproj_w; };
  struct DecP   { const float *dw1_b,*dw1_w,*dw2_b,*dw2_w,*g1,*g2; MambaP mb[2];
                  const float *n1_b,*n1_w,*n2_b,*n2_w,*pw11_b,*pw11_w,*pw12_b,*pw12_w,
                              *pw21_b,*pw21_w,*pw22_b,*pw22_w; };
  struct BlkP   { const float *dw_b,*dw_w,*gamma,*ln_b,*ln_w,*pw1_b,*pw1_w,*pw2_b,*pw2_w; };

  DecP dec[6];
  for (int l = 0; l < 6; l++) {
    DecP& p = dec[l];
    p.dw1_b = nxt(); p.dw1_w = nxt(); p.dw2_b = nxt(); p.dw2_w = nxt();
    p.g1 = nxt(); p.g2 = nxt();
    for (int m = 0; m < 2; m++) {
      MambaP& q = p.mb[m];
      q.A_log = nxt(); q.D = nxt(); q.conv_b = nxt(); q.conv_w = nxt();
      q.dt_b = nxt(); q.dt_w = nxt(); q.in_w = nxt(); q.out_w = nxt(); q.xproj_w = nxt();
    }
    p.n1_b = nxt(); p.n1_w = nxt(); p.n2_b = nxt(); p.n2_w = nxt();
    p.pw11_b = nxt(); p.pw11_w = nxt(); p.pw12_b = nxt(); p.pw12_w = nxt();
    p.pw21_b = nxt(); p.pw21_w = nxt(); p.pw22_b = nxt(); p.pw22_w = nxt();
  }
  const float* dec_norm_b = nxt();
  const float* dec_norm_w = nxt();
  static const int ENCD[3] = {512, 1024, 512};
  static const int ENCDEP[3] = {2, 6, 2};
  BlkP blk[3][6];
  for (int s = 0; s < 3; s++)
    for (int b = 0; b < ENCDEP[s]; b++) {
      BlkP& q = blk[s][b];
      q.dw_b = nxt(); q.dw_w = nxt(); q.gamma = nxt(); q.ln_b = nxt(); q.ln_w = nxt();
      q.pw1_b = nxt(); q.pw1_w = nxt(); q.pw2_b = nxt(); q.pw2_w = nxt();
    }
  const float *t1_b = nxt(), *t1_w = nxt(), *t2_b = nxt(), *t2_w = nxt();
  const float* tgt_embed = nxt();

  // ---- workspace bump allocator ----
  size_t off = 0;
  auto alloc = [&](size_t nflt) -> float* {
    float* p = (float*)((char*)d_ws + off);
    off += ((nflt * sizeof(float) + 255) & ~(size_t)255);
    return p;
  };
  float* qpos  = alloc((size_t)MTOK * 512);
  float* maskf = alloc(MTOK);
  float* x0    = alloc((size_t)MTOK * 1024);
  float* x1    = alloc((size_t)MTOK * 1024);
  float* bufA  = alloc((size_t)MTOK * 1024);
  float* bufB  = alloc((size_t)MTOK * 1024);
  float* bufC  = alloc((size_t)MTOK * 1024);
  float* big   = alloc((size_t)MTOK * 4096);
  float* mem   = alloc((size_t)MTOK * 512);
  float* tgt   = alloc((size_t)MTOK * 512);
  float* chk   = alloc((size_t)MTOK * 512);
  float* xz    = alloc((size_t)MTOK * 512);
  float* xcv   = alloc((size_t)MTOK * 256);
  float* xdb   = alloc((size_t)MTOK * 40);
  float* dtb   = alloc((size_t)MTOK * 256);
  float* ysc   = alloc((size_t)MTOK * 256);
  float* m1o   = alloc((size_t)MTOK * 128);

  auto ge = [&](const float* A, int lda, const float* W, int K, const float* bias,
                const float* mk, float* C, int ldc, int N, int act) {
    dim3 g((unsigned)((N + 63) / 64), (unsigned)(MTOK / 128));
    gemm_f16_wmma<<<g, 128, 0, stream>>>(A, lda, W, K, bias, mk, C, ldc, N, act);
  };
  auto conv = [&](const float* X, int ldx, const float* Wt, const float* Bv, const float* mk,
                  float* Y, int ldy, int C, int taps, int padL, int silu) {
    int tot = MTOK * C;
    dwconv_kernel<<<(tot + 255) / 256, 256, 0, stream>>>(X, ldx, Wt, Bv, mk, Y, ldy,
                                                          SEQN, C, taps, padL, silu, tot);
  };
  auto ln = [&](const float* X, int ldx, const float* w, const float* b,
                float* Y, int ldy, int C, float eps) {
    ln_kernel<<<MTOK, 256, 0, stream>>>(X, ldx, w, b, Y, ldy, C, eps);
  };
  auto eadd = [&](const float* a, const float* b, float* o, int n) {
    add_kernel<<<(n + 255) / 256, 256, 0, stream>>>(a, b, o, n);
  };
  auto resg = [&](const float* res, const float* g, const float* y, float* o, int n, int C) {
    resg_kernel<<<(n + 255) / 256, 256, 0, stream>>>(res, g, y, o, n, C);
  };
  auto mamba = [&](const MambaP& q, const float* xin, int ldin, float* out, int ldout) {
    ge(xin, ldin, q.in_w, 128, nullptr, nullptr, xz, 512, 512, ACT_NONE);       // in_proj
    conv(xz, 512, q.conv_w, q.conv_b, nullptr, xcv, 256, 256, 4, 3, 1);         // causal conv4 + silu
    ge(xcv, 256, q.xproj_w, 256, nullptr, nullptr, xdb, 40, 40, ACT_NONE);      // x_proj
    dtproj_kernel<<<MTOK, 256, 0, stream>>>(xdb, q.dt_w, q.dt_b, dtb);          // softplus(dt)
    scan_kernel<<<BATCHN, 256, 0, stream>>>(dtb, xdb, xcv, xz + 256, 512,
                                            q.A_log, q.D, ysc, SEQN);           // SSM + silu(z) gate
    ge(ysc, 256, q.out_w, 256, nullptr, nullptr, out, ldout, 128, ACT_NONE);    // out_proj
  };

  // ---- preprocessing ----
  maskf_kernel<<<(MTOK + 255) / 256, 256, 0, stream>>>(maskb, maskf, MTOK);
  t_bcl_blc<<<(MTOK * 512 + 255) / 256, 256, 0, stream>>>(src_bcl,  x0,   512, SEQN, MTOK * 512);
  t_bcl_blc<<<(MTOK * 512 + 255) / 256, 256, 0, stream>>>(qpos_bcl, qpos, 512, SEQN, MTOK * 512);

  // ---- encoder ----
  float* x = x0; float* xalt = x1;
  for (int s = 0; s < 3; s++) {
    int dim = ENCD[s];
    if (s == 1) { ge(x,  512, t1_w,  512, t1_b, maskf, xalt, 1024, 1024, ACT_NONE);
                  float* t_ = x; x = xalt; xalt = t_; }
    if (s == 2) { ge(x, 1024, t2_w, 1024, t2_b, maskf, xalt,  512,  512, ACT_NONE);
                  float* t_ = x; x = xalt; xalt = t_; }
    for (int b = 0; b < ENCDEP[s]; b++) {
      const BlkP& q = blk[s][b];
      conv(x, dim, q.dw_w, q.dw_b, maskf, bufA, dim, dim, 7, 3, 0);
      ln(bufA, dim, q.ln_w, q.ln_b, bufA, dim, dim, 1e-6f);
      ge(bufA, dim, q.pw1_w, dim, q.pw1_b, nullptr, big, 4 * dim, 4 * dim, ACT_GELU);
      ge(big, 4 * dim, q.pw2_w, 4 * dim, q.pw2_b, nullptr, bufB, dim, dim, ACT_NONE);
      resg(x, q.gamma, bufB, x, MTOK * dim, dim);
    }
  }
  copy_kernel<<<(MTOK * 512 + 255) / 256, 256, 0, stream>>>(x, mem, MTOK * 512);

  // ---- decoder ----
  bcast_kernel<<<(MTOK * 512 + 255) / 256, 256, 0, stream>>>(tgt_embed, tgt, MTOK * 512, SEQN * 512);
  const int NE = MTOK * 512;
  for (int l = 0; l < 6; l++) {
    const DecP& p = dec[l];
    for (int i = 0; i < 4; i++) {                       // 4 channel chunks x 2 mamba
      mamba(p.mb[0], tgt + i * 128, 512, m1o, 128);
      mamba(p.mb[1], m1o, 128, chk + i * 128, 512);
    }
    eadd(tgt, chk, tgt, NE);                            // tgt += concat(chunks)
    eadd(tgt, qpos, bufA, NE);                          // tgt2 = tgt + query_pos
    conv(bufA, 512, p.dw1_w, p.dw1_b, maskf, bufB, 512, 512, 9, 4, 0);
    ln(bufB, 512, p.n1_w, p.n1_b, bufB, 512, 512, 1e-6f);
    ge(bufB, 512, p.pw11_w, 512, p.pw11_b, nullptr, big, 2048, 2048, ACT_GELU);
    ge(big, 2048, p.pw12_w, 2048, p.pw12_b, nullptr, bufC, 512, 512, ACT_NONE);
    resg(tgt, p.g1, bufC, tgt, NE, 512);                // tgt += g1 * mlp
    eadd(tgt, mem, bufA, NE);                           // tgt2 = tgt + memory
    conv(bufA, 512, p.dw2_w, p.dw2_b, maskf, bufB, 512, 512, 9, 4, 0);
    eadd(bufB, tgt, bufB, NE);                          // tgt2 += tgt
    ln(bufB, 512, p.n2_w, p.n2_b, bufB, 512, 512, 1e-6f);
    ge(bufB, 512, p.pw21_w, 512, p.pw21_b, nullptr, big, 2048, 2048, ACT_GELU);
    ge(big, 2048, p.pw22_w, 2048, p.pw22_b, nullptr, bufC, 512, 512, ACT_NONE);
    resg(bufB, p.g2, bufC, tgt, NE, 512);               // tgt = LN_out + g2 * mlp
  }

  // ---- outputs: [hs (B,QN,d)] [memory (B,C,L)] [mask (B,1,L)] ----
  float* outp = (float*)d_out;
  ln(tgt, 512, dec_norm_w, dec_norm_b, outp, 512, 512, 1e-5f);
  t_blc_bcl<<<(MTOK * 512 + 255) / 256, 256, 0, stream>>>(mem, outp + (size_t)MTOK * 512,
                                                          512, SEQN, MTOK * 512);
  copy_kernel<<<(MTOK + 255) / 256, 256, 0, stream>>>(maskf, outp + (size_t)MTOK * 1024, MTOK);
}
